// TransformerModel_89154931130431
// MI455X (gfx1250) — compile-verified, matrix operand order checked by reading, not run on hardware
//
#include <hip/hip_runtime.h>
#include <hip/hip_bf16.h>

// ---------------------------------------------------------------------------
// Model dims (match reference)
// ---------------------------------------------------------------------------
#define NPART   16384
#define NGRAPH  128
#define MAXL    256
#define DMODEL  128
#define NHEAD   4
#define DHEAD   32
#define NBLK    10
#define INDIM   16
#define OUTDIM  4
#define LN_EPS  1e-5f

typedef __attribute__((ext_vector_type(16))) __bf16 v16bf;
typedef __attribute__((ext_vector_type(8)))  float  v8f;

union BF8 { uint4 q; __bf16 b[8]; };   // 8 bf16 <-> one 128-bit load/store

// ---------------------------------------------------------------------------
// WMMA helpers (gfx1250: V_WMMA_F32_16X16X32_BF16, wave32)
// ---------------------------------------------------------------------------
__device__ __forceinline__ v8f wmma_bf16(v16bf a, v16bf b, v8f c) {
  // (neg_a, A, neg_b, B, c_mod, C, reuse_a, reuse_b)
  return __builtin_amdgcn_wmma_f32_16x16x32_bf16(false, a, false, b, (short)0, c,
                                                 false, false);
}

__device__ __forceinline__ v8f splat8(float x) {
  v8f r;
#pragma unroll
  for (int i = 0; i < 8; ++i) r[i] = x;
  return r;
}

// Load a 16x32 bf16 operand tile in the documented 16-bit A/B VGPR layout.
// `base` is "N-major" (row index = the 16-wide operand dimension, contiguous
// along K).  Lane l: row = r0 + (l&15); elements j:
//   k = k0 + j + 8*((j>>3) + (l>>4))  -> two contiguous 8-element (16B) loads.
__device__ __forceinline__ v16bf load_opnd(const __bf16* base, int ld, int r0, int k0) {
  const int lane = threadIdx.x & 31;
  const int row  = r0 + (lane & 15);
  const int kh   = (lane >> 4) << 3;   // 0 or 8
  const __bf16* p = base + row * ld + k0 + kh;
  union { v16bf v; uint4 q[2]; } u;
  u.q[0] = *(const uint4*)(p);        // j = 0..7
  u.q[1] = *(const uint4*)(p + 16);   // j = 8..15
  return u.v;
}

// ---------------------------------------------------------------------------
// k0: transpose+convert weights fp32 [B,K,N] -> bf16 [B,N,K]
// ---------------------------------------------------------------------------
__global__ void wconv_kernel(const float* __restrict__ src, __bf16* __restrict__ dst,
                             int B, int K, int N) {
  long total = (long)B * K * N;
  for (long i = (long)blockIdx.x * blockDim.x + threadIdx.x; i < total;
       i += (long)gridDim.x * blockDim.x) {
    long b = i / ((long)K * N);
    long r = (i / N) % K;
    long c = i % N;
    dst[(b * N + c) * K + r] = (__bf16)src[i];
  }
}

// ---------------------------------------------------------------------------
// k1: BatchNorm batch statistics (training-mode, biased var) over [N,16]
// ---------------------------------------------------------------------------
__global__ void bnstats_kernel(const float* __restrict__ xs, const float* __restrict__ xv,
                               float* __restrict__ mu, float* __restrict__ var) {
  const int col = threadIdx.x & 15;
  const int grp = threadIdx.x >> 4;
  float s = 0.f, q = 0.f;
  for (int n = grp; n < NPART; n += 16) {
    float v = (col < 13) ? xs[n * 13 + col] : xv[n * 3 + (col - 13)];
    s += v; q += v * v;
  }
  __shared__ float S[256], Q[256];
  S[threadIdx.x] = s; Q[threadIdx.x] = q;
  __syncthreads();
  if (grp == 0) {
    for (int g2 = 1; g2 < 16; ++g2) { s += S[g2 * 16 + col]; q += Q[g2 * 16 + col]; }
    float m = s / (float)NPART;
    mu[col]  = m;
    var[col] = q / (float)NPART - m * m;
  }
}

// ---------------------------------------------------------------------------
// k2: segment offsets via binary search (batch_idx is sorted)
// ---------------------------------------------------------------------------
__global__ void offsets_kernel(const int* __restrict__ ids, int* __restrict__ off) {
  int g = threadIdx.x;
  if (g > NGRAPH) return;
  int lo = 0, hi = NPART;
  while (lo < hi) { int mid = (lo + hi) >> 1; if (ids[mid] < g) lo = mid + 1; else hi = mid; }
  off[g] = lo;
}

// ---------------------------------------------------------------------------
// k3: BN apply + input embed 16 -> 128, writes bf16 rows (vector stores)
// ---------------------------------------------------------------------------
__global__ void embed_kernel(const float* __restrict__ xs, const float* __restrict__ xv,
                             const float* __restrict__ mu, const float* __restrict__ var,
                             const float* __restrict__ bng, const float* __restrict__ bnb,
                             const float* __restrict__ w_in, const float* __restrict__ b_in,
                             __bf16* __restrict__ h) {
  int n = blockIdx.x * blockDim.x + threadIdx.x;
  if (n >= NPART) return;
  float x[INDIM];
#pragma unroll
  for (int j = 0; j < 13; ++j) x[j] = xs[n * 13 + j];
#pragma unroll
  for (int j = 0; j < 3; ++j) x[13 + j] = xv[n * 3 + j];
#pragma unroll
  for (int j = 0; j < INDIM; ++j)
    x[j] = (x[j] - mu[j]) * rsqrtf(var[j] + LN_EPS) * bng[j] + bnb[j];
  __bf16* hr = h + (long)n * DMODEL;
  for (int d = 0; d < DMODEL; d += 8) {
    BF8 o;
#pragma unroll
    for (int i = 0; i < 8; ++i) {
      float a = b_in[d + i];
#pragma unroll
      for (int j = 0; j < INDIM; ++j) a += x[j] * w_in[j * DMODEL + d + i];
      o.b[i] = (__bf16)a;
    }
    *(uint4*)(hr + d) = o.q;
  }
}

// ---------------------------------------------------------------------------
// LayerNorm over the LDS residual; one row per thread, 128-bit LDS accesses
// ---------------------------------------------------------------------------
__device__ __forceinline__ void layer_norm_rows(const __bf16* X, __bf16* Y,
                                                const float* gamma, const float* beta) {
  const int row = threadIdx.x;
  const __bf16* xr = X + row * DMODEL;
  float s = 0.f, q = 0.f;
#pragma unroll 4
  for (int d = 0; d < DMODEL; d += 8) {
    BF8 u; u.q = *(const uint4*)(xr + d);
#pragma unroll
    for (int i = 0; i < 8; ++i) { float v = (float)u.b[i]; s += v; q += v * v; }
  }
  const float mean = s * (1.f / DMODEL);
  const float varr = q * (1.f / DMODEL) - mean * mean;
  const float rs = rsqrtf(varr + LN_EPS);
  __bf16* yr = Y + row * DMODEL;
#pragma unroll 4
  for (int d = 0; d < DMODEL; d += 8) {
    BF8 u; u.q = *(const uint4*)(xr + d);
    BF8 o;
#pragma unroll
    for (int i = 0; i < 8; ++i)
      o.b[i] = (__bf16)(((float)u.b[i] - mean) * rs * gamma[d + i] + beta[d + i]);
    *(uint4*)(yr + d) = o.q;
  }
}

// ---------------------------------------------------------------------------
// k4: whole 10-block transformer, one workgroup per graph, all-LDS resident.
// Dynamic LDS layout (bytes):
//   X16  @      0  [256][128] bf16 residual         65536
//   Y16  @  65536  [256][128] bf16 LN output        65536
//   AO   @ 131072  [256][128] bf16 attn-out / MLP   65536
//   Qh   @ 196608  [256][32]  bf16                  16384
//   Kh   @ 212992  [256][32]  bf16                  16384
//   VhT  @ 229376  [32][256]  bf16 (transposed V)   16384
//   Pst  @ 245760  8 waves x [16][32] bf16           8192   -> 253952 total
// ---------------------------------------------------------------------------
#define SMEM_BYTES 253952

__global__ __launch_bounds__(256, 1)
void part_main_kernel(const __bf16* __restrict__ h, const int* __restrict__ off,
                      const __bf16* __restrict__ WqkvT, const float* __restrict__ b_qkv,
                      const __bf16* __restrict__ WoT,   const float* __restrict__ b_o,
                      const __bf16* __restrict__ Wm1T,  const float* __restrict__ b_m1,
                      const __bf16* __restrict__ Wm2T,  const float* __restrict__ b_m2,
                      const float* __restrict__ ln1g, const float* __restrict__ ln1b,
                      const float* __restrict__ ln2g, const float* __restrict__ ln2b,
                      const float* __restrict__ lnfg, const float* __restrict__ lnfb,
                      const float* __restrict__ w_out, const float* __restrict__ b_out,
                      float* __restrict__ out) {
  extern __shared__ char smem[];
  __bf16* X16 = (__bf16*)(smem);
  __bf16* Y16 = (__bf16*)(smem + 65536);
  __bf16* AO  = (__bf16*)(smem + 131072);
  __bf16* Qh  = (__bf16*)(smem + 196608);
  __bf16* Kh  = (__bf16*)(smem + 212992);
  __bf16* VhT = (__bf16*)(smem + 229376);
  __bf16* Pst = (__bf16*)(smem + 245760);

  const int gid  = blockIdx.x;
  const int tid  = threadIdx.x;
  const int w    = tid >> 5;
  const int lane = tid & 31;
  const int lo   = lane & 15;
  const int hi8  = (lane >> 4) << 3;

  const int off0 = off[gid];
  const int cnt  = off[gid + 1] - off0;

  // --- load residual rows (contiguous segment; pure 16B vector copy) ---
  {
    __bf16* xr = X16 + tid * DMODEL;
    if (tid < cnt) {
      const __bf16* hr = h + (long)(off0 + tid) * DMODEL;
#pragma unroll
      for (int d = 0; d < DMODEL; d += 8)
        *(uint4*)(xr + d) = *(const uint4*)(hr + d);
    } else {
      uint4 z = {0u, 0u, 0u, 0u};
#pragma unroll
      for (int d = 0; d < DMODEL; d += 8) *(uint4*)(xr + d) = z;
    }
  }
  __syncthreads();

  const float iscale = 0.17677669529663687f;  // 1/sqrt(DHEAD)

  for (int blk = 0; blk < NBLK; ++blk) {
    const __bf16* Wq = WqkvT + (long)blk * 384 * 128;
    const __bf16* Wo = WoT   + (long)blk * 128 * 128;
    const __bf16* W1 = Wm1T  + (long)blk * 512 * 128;
    const __bf16* W2 = Wm2T  + (long)blk * 128 * 512;

    // prefetch next block's weight panels into the cache hierarchy
    if (blk + 1 < NBLK) {
      __builtin_prefetch(WqkvT + (long)(blk + 1) * 384 * 128 + tid * 192, 0, 1);
      __builtin_prefetch(WoT   + (long)(blk + 1) * 128 * 128 + tid * 64,  0, 1);
      __builtin_prefetch(Wm1T  + (long)(blk + 1) * 512 * 128 + tid * 256, 0, 1);
      __builtin_prefetch(Wm2T  + (long)(blk + 1) * 128 * 512 + tid * 256, 0, 1);
    }

    // ---------------- LN1 ----------------
    layer_norm_rows(X16, Y16, ln1g + blk * 128, ln1b + blk * 128);
    __syncthreads();

    // ---------------- attention, one head at a time ----------------
    for (int hh = 0; hh < NHEAD; ++hh) {
      // --- Q/K/V for this head: [256,128] x [128,96] via WMMA.
      // Register-blocked: each wave owns 2 m-tiles; for each, all 6 n-tiles
      // are held in registers so every A operand is loaded once per k-step.
#pragma unroll
      for (int mi = 0; mi < 2; ++mi) {
        const int mt = w + mi * 8;
        v8f c[6];
#pragma unroll
        for (int nt = 0; nt < 6; ++nt) {
          const int wrow0 = (nt >> 1) * 128 + hh * 32 + (nt & 1) * 16;
          c[nt] = splat8(b_qkv[blk * 384 + wrow0 + lo]);
        }
#pragma unroll
        for (int kk = 0; kk < 4; ++kk) {
          v16bf a = load_opnd(Y16, 128, mt * 16, kk * 32);
#pragma unroll
          for (int nt = 0; nt < 6; ++nt) {
            const int wrow0 = (nt >> 1) * 128 + hh * 32 + (nt & 1) * 16;
            v16bf b = load_opnd(Wq, 128, wrow0, kk * 32);
            c[nt] = wmma_bf16(a, b, c[nt]);
          }
        }
#pragma unroll
        for (int nt = 0; nt < 6; ++nt) {
          const int sec  = nt >> 1;              // 0=Q 1=K 2=V (compile-time)
          const int half = nt & 1;
          __bf16* sbase = (sec == 0) ? Qh : ((sec == 1) ? Kh : VhT);
          const int rs = (sec == 2) ? 1 : 32;
          const int cs = (sec == 2) ? 256 : 1;
          const int cb = (half * 16 + lo) * cs;
#pragma unroll
          for (int e = 0; e < 8; ++e) {
            const int r = mt * 16 + e + hi8;
            sbase[r * rs + cb] = (__bf16)c[nt][e];
          }
        }
      }
      __syncthreads();

      // --- flash attention: 2 query tiles per wave, keys in chunks of 32 ---
      __bf16* Pw = Pst + w * (16 * 32);
      for (int qt = w; qt < 16; qt += 8) {
        v16bf aq = load_opnd(Qh, 32, qt * 16, 0);   // K-dim = DHEAD = 32
        float mrow[8], lrow[8];
        v8f o0 = splat8(0.f), o1 = splat8(0.f);
#pragma unroll
        for (int e = 0; e < 8; ++e) { mrow[e] = -1e30f; lrow[e] = 0.f; }

        for (int kc = 0; kc < MAXL; kc += 32) {
          v16bf bk0 = load_opnd(Kh, 32, kc, 0);       // keys are the N dim
          v16bf bk1 = load_opnd(Kh, 32, kc + 16, 0);
          v8f z = splat8(0.f);
          v8f s0 = wmma_bf16(aq, bk0, z);
          v8f s1 = wmma_bf16(aq, bk1, z);
          const float bias0 = (kc + lo)      < cnt ? 0.f : -1e9f;
          const float bias1 = (kc + 16 + lo) < cnt ? 0.f : -1e9f;
#pragma unroll
          for (int e = 0; e < 8; ++e) {
            float v0 = s0[e] * iscale + bias0;
            float v1 = s1[e] * iscale + bias1;
            // row max over the 32-key chunk (16-lane half-wave reduce)
            float mx = fmaxf(v0, v1);
            mx = fmaxf(mx, __shfl_xor(mx, 1, 32));
            mx = fmaxf(mx, __shfl_xor(mx, 2, 32));
            mx = fmaxf(mx, __shfl_xor(mx, 4, 32));
            mx = fmaxf(mx, __shfl_xor(mx, 8, 32));
            const float mnew  = fmaxf(mrow[e], mx);
            const float alpha = __expf(mrow[e] - mnew);
            const float p0 = __expf(v0 - mnew);
            const float p1 = __expf(v1 - mnew);
            float ps = p0 + p1;
            ps += __shfl_xor(ps, 1, 32);
            ps += __shfl_xor(ps, 2, 32);
            ps += __shfl_xor(ps, 4, 32);
            ps += __shfl_xor(ps, 8, 32);
            lrow[e] = lrow[e] * alpha + ps;
            o0[e] *= alpha; o1[e] *= alpha;
            mrow[e] = mnew;
            // stage P (C-layout -> A-layout) through per-wave LDS
            const int pr = e + hi8;
            Pw[pr * 32 + lo]      = (__bf16)p0;
            Pw[pr * 32 + 16 + lo] = (__bf16)p1;
          }
          // same-wave LDS ordering: no workgroup barrier needed
          v16bf ap  = load_opnd(Pw, 32, 0, 0);
          v16bf bv0 = load_opnd(VhT, 256, 0,  kc);
          v16bf bv1 = load_opnd(VhT, 256, 16, kc);
          o0 = wmma_bf16(ap, bv0, o0);
          o1 = wmma_bf16(ap, bv1, o1);
        }
#pragma unroll
        for (int e = 0; e < 8; ++e) {
          const int r = qt * 16 + e + hi8;
          const float inv = 1.f / lrow[e];
          AO[r * 128 + hh * 32 + lo]      = (__bf16)(o0[e] * inv);
          AO[r * 128 + hh * 32 + 16 + lo] = (__bf16)(o1[e] * inv);
        }
      }
      __syncthreads();
    }

    // ---------------- O-projection + residual (register-blocked 8 n-tiles) --
#pragma unroll
    for (int mi = 0; mi < 2; ++mi) {
      const int mt = w + mi * 8;
      v8f c[8];
#pragma unroll
      for (int nt = 0; nt < 8; ++nt)
        c[nt] = splat8(b_o[blk * 128 + nt * 16 + lo]);
#pragma unroll
      for (int kk = 0; kk < 4; ++kk) {
        v16bf a = load_opnd(AO, 128, mt * 16, kk * 32);
#pragma unroll
        for (int nt = 0; nt < 8; ++nt) {
          v16bf b = load_opnd(Wo, 128, nt * 16, kk * 32);
          c[nt] = wmma_bf16(a, b, c[nt]);
        }
      }
#pragma unroll
      for (int nt = 0; nt < 8; ++nt)
#pragma unroll
        for (int e = 0; e < 8; ++e) {
          const int idx = (mt * 16 + e + hi8) * 128 + nt * 16 + lo;
          X16[idx] = (__bf16)((float)X16[idx] + c[nt][e]);
        }
    }
    __syncthreads();

    // ---------------- LN2 + MLP (hidden in 4 chunks of 128) ----------------
    layer_norm_rows(X16, Y16, ln2g + blk * 128, ln2b + blk * 128);
    __syncthreads();

    for (int cc = 0; cc < 4; ++cc) {
      // hidden chunk = gelu(Y @ W1[:, cc*128 : +128]) -> AO (reused)
#pragma unroll
      for (int mi = 0; mi < 2; ++mi) {
        const int mt = w + mi * 8;
        v8f c[8];
#pragma unroll
        for (int nt = 0; nt < 8; ++nt)
          c[nt] = splat8(b_m1[blk * 512 + cc * 128 + nt * 16 + lo]);
#pragma unroll
        for (int kk = 0; kk < 4; ++kk) {
          v16bf a = load_opnd(Y16, 128, mt * 16, kk * 32);
#pragma unroll
          for (int nt = 0; nt < 8; ++nt) {
            v16bf b = load_opnd(W1, 128, cc * 128 + nt * 16, kk * 32);
            c[nt] = wmma_bf16(a, b, c[nt]);
          }
        }
#pragma unroll
        for (int nt = 0; nt < 8; ++nt)
#pragma unroll
          for (int e = 0; e < 8; ++e) {
            const float v = c[nt][e];
            const float t3 = v + 0.044715f * v * v * v;
            const float ge = 0.5f * v * (1.f + tanhf(0.7978845608028654f * t3));
            AO[(mt * 16 + e + hi8) * 128 + nt * 16 + lo] = (__bf16)ge;
          }
      }
      __syncthreads();
      // X += hidden_chunk @ W2[cc*128 : , :]
#pragma unroll
      for (int mi = 0; mi < 2; ++mi) {
        const int mt = w + mi * 8;
        v8f c[8];
#pragma unroll
        for (int nt = 0; nt < 8; ++nt)
          c[nt] = splat8((cc == 0) ? b_m2[blk * 128 + nt * 16 + lo] : 0.f);
#pragma unroll
        for (int kk = 0; kk < 4; ++kk) {
          v16bf a = load_opnd(AO, 128, mt * 16, kk * 32);
#pragma unroll
          for (int nt = 0; nt < 8; ++nt) {
            v16bf b = load_opnd(W2, 512, nt * 16, cc * 128 + kk * 32);
            c[nt] = wmma_bf16(a, b, c[nt]);
          }
        }
#pragma unroll
        for (int nt = 0; nt < 8; ++nt)
#pragma unroll
          for (int e = 0; e < 8; ++e) {
            const int idx = (mt * 16 + e + hi8) * 128 + nt * 16 + lo;
            X16[idx] = (__bf16)((float)X16[idx] + c[nt][e]);
          }
      }
      __syncthreads();
    }
  }

  // ---------------- final LN + tiny output head + sigmoid ----------------
  layer_norm_rows(X16, Y16, lnfg, lnfb);
  __syncthreads();
  {
    const int row = tid;
    if (row < cnt) {
      const __bf16* yr = Y16 + row * DMODEL;
      float a0 = b_out[0], a1 = b_out[1], a2 = b_out[2], a3 = b_out[3];
#pragma unroll 4
      for (int d = 0; d < DMODEL; d += 8) {
        BF8 u; u.q = *(const uint4*)(yr + d);
#pragma unroll
        for (int i = 0; i < 8; ++i) {
          const float yv = (float)u.b[i];
          a0 += yv * w_out[(d + i) * 4 + 0];
          a1 += yv * w_out[(d + i) * 4 + 1];
          a2 += yv * w_out[(d + i) * 4 + 2];
          a3 += yv * w_out[(d + i) * 4 + 3];
        }
      }
      a3 = 1.f / (1.f + __expf(-a3));
      float* o = out + (long)(off0 + row) * 4;
      o[0] = a0; o[1] = a1; o[2] = a2; o[3] = a3;
    }
  }
}

// ---------------------------------------------------------------------------
// Host launcher
// ---------------------------------------------------------------------------
extern "C" void kernel_launch(void* const* d_in, const int* in_sizes, int n_in,
                              void* d_out, int out_size, void* d_ws, size_t ws_size,
                              hipStream_t stream) {
  (void)in_sizes; (void)n_in; (void)out_size; (void)ws_size;
  const float* in_scalar = (const float*)d_in[0];
  const float* in_v      = (const float*)d_in[1];
  const int*   batch_idx = (const int*)  d_in[2];
  const float* bn_g      = (const float*)d_in[3];
  const float* bn_b      = (const float*)d_in[4];
  const float* w_in      = (const float*)d_in[5];
  const float* b_in      = (const float*)d_in[6];
  const float* ln1_g     = (const float*)d_in[7];
  const float* ln1_b     = (const float*)d_in[8];
  const float* w_qkv     = (const float*)d_in[9];
  const float* b_qkv     = (const float*)d_in[10];
  const float* w_o       = (const float*)d_in[11];
  const float* b_o       = (const float*)d_in[12];
  const float* ln2_g     = (const float*)d_in[13];
  const float* ln2_b     = (const float*)d_in[14];
  const float* w_m1      = (const float*)d_in[15];
  const float* b_m1      = (const float*)d_in[16];
  const float* w_m2      = (const float*)d_in[17];
  const float* b_m2      = (const float*)d_in[18];
  const float* lnf_g     = (const float*)d_in[19];
  const float* lnf_b     = (const float*)d_in[20];
  const float* w_out     = (const float*)d_in[21];
  const float* b_out     = (const float*)d_in[22];
  float* out = (float*)d_out;

  // workspace layout
  char* ws = (char*)d_ws;
  int*   off  = (int*)(ws + 0);                 // 129 ints
  float* mu   = (float*)(ws + 1024);            // 16 floats
  float* var  = (float*)(ws + 1152);            // 16 floats
  __bf16* h     = (__bf16*)(ws + 4096);         // N*128 bf16 = 4 MB
  __bf16* WqkvT = (__bf16*)(ws + 4198400);      // 10*384*128 bf16
  __bf16* WoT   = (__bf16*)(ws + 5181440);      // 10*128*128 bf16
  __bf16* Wm1T  = (__bf16*)(ws + 5509120);      // 10*512*128 bf16
  __bf16* Wm2T  = (__bf16*)(ws + 6819840);      // 10*128*512 bf16 (ends 8130560)

  // weight convert + transpose (bf16, Dout-major)
  wconv_kernel<<<480, 256, 0, stream>>>(w_qkv, WqkvT, NBLK, 128, 384);
  wconv_kernel<<<160, 256, 0, stream>>>(w_o,   WoT,   NBLK, 128, 128);
  wconv_kernel<<<640, 256, 0, stream>>>(w_m1,  Wm1T,  NBLK, 128, 512);
  wconv_kernel<<<640, 256, 0, stream>>>(w_m2,  Wm2T,  NBLK, 512, 128);

  // BN stats, segment offsets, embed
  bnstats_kernel<<<1, 256, 0, stream>>>(in_scalar, in_v, mu, var);
  offsets_kernel<<<1, 256, 0, stream>>>(batch_idx, off);
  embed_kernel<<<(NPART + 255) / 256, 256, 0, stream>>>(
      in_scalar, in_v, mu, var, bn_g, bn_b, w_in, b_in, h);

  // whole transformer: one workgroup per graph, 248 KB dynamic LDS
  part_main_kernel<<<NGRAPH, 256, SMEM_BYTES, stream>>>(
      h, off, WqkvT, b_qkv, WoT, b_o, Wm1T, b_m1, Wm2T, b_m2,
      ln1_g, ln1_b, ln2_g, ln2_b, lnf_g, lnf_b, w_out, b_out, out);
}
